// NTXentLoss_84499186581515
// MI455X (gfx1250) — matrix operand order repaired and checked by reading, not compiled, not run on hardware
//
#include <hip/hip_runtime.h>
#include <hip/hip_fp16.h>

// ---------------------------------------------------------------------------
// NT-Xent loss, MI455X (gfx1250), wave32 + v_wmma_f32_16x16x32_f16
//   x: [4096, 512] f32  ->  scalar loss
//   sim = (x/||x||) @ (x/||x||)^T ; logits = sim/0.5 with -inf diag
//   loss = mean_i( logsumexp_i - logit[i, i^1] )
// Bounded-logit trick: |sim|<=1 -> logits in [-2,2], so fixed max=2.0 is a
// numerically safe softmax shift: lse_i = 2 + log(sum_j exp(logit_ij - 2)).
// ---------------------------------------------------------------------------

typedef __attribute__((ext_vector_type(16))) _Float16 v16h;
typedef __attribute__((ext_vector_type(8)))  float    v8f;

#define N_ROWS   4096
#define DIM      512
#define N_TILES  (N_ROWS / 16)   // 256
#define INV_TEMP 2.0f
#define EPS_N    1e-8f
#define PANEL    (16 * DIM)      // halves per 16-row panel = 8192

union FragU { v16h v; uint4 q[2]; };

static __device__ __forceinline__ v8f wmma_f16(v16h a, v16h b, v8f c) {
    return __builtin_amdgcn_wmma_f32_16x16x32_f16(
        false, a, false, b, (short)0, c, false, false);
}

// ---------------------------------------------------------------------------
// Kernel 1: per-row L2 normalize, f32 -> f16 (row-major [4096][512])
// ---------------------------------------------------------------------------
__global__ void __launch_bounds__(256)
normalize_kernel(const float* __restrict__ x, _Float16* __restrict__ xh) {
    const int row = blockIdx.x;
    const int tid = threadIdx.x;
    const float* xr = x + row * DIM;

    float2 v = ((const float2*)xr)[tid];
    float ss = v.x * v.x + v.y * v.y;
    #pragma unroll
    for (int off = 16; off > 0; off >>= 1) ss += __shfl_xor(ss, off);

    __shared__ float wsum[8];
    if ((tid & 31) == 0) wsum[tid >> 5] = ss;
    __syncthreads();

    float tot = 0.f;
    #pragma unroll
    for (int w = 0; w < 8; ++w) tot += wsum[w];

    const float inv = 1.0f / fmaxf(sqrtf(tot), EPS_N);

    union { _Float16 h[2]; unsigned u; } p;
    p.h[0] = (_Float16)(v.x * inv);
    p.h[1] = (_Float16)(v.y * inv);
    ((unsigned*)xh)[row * (DIM / 2) + tid] = p.u;
}

// ---------------------------------------------------------------------------
// Kernel 2: WMMA strip GEMM, fixed-shift softmax accumulation.
// Block m_tile owns rows [m_tile*16, +16). 8 waves; wave w sweeps tile pairs
// p = w + 8t (tiles 2p, 2p+1) with resident A fragments and a quad-buffered
// B pipeline at prefetch distance 2 that rolls across tile-pair boundaries
// (16 % 4 == 0 keeps buffer indices loop-invariant).
// ---------------------------------------------------------------------------
__global__ void __launch_bounds__(256)
ntxent_strip_kernel(const _Float16* __restrict__ xh, float* __restrict__ partial) {
    const int m_tile = blockIdx.x;           // 0..255
    const int tid  = threadIdx.x;
    const int wave = tid >> 5;
    const int lane = tid & 31;
    const int l    = lane & 15;              // lane-in-group (column / row id)
    const int hi   = lane >> 4;              // half-wave select
    const int r0   = m_tile * 16;

    __shared__ float red_s[8][16];
    __shared__ float t_sh[16];

    // ---- resident A fragments: row r0+l, contiguous K packing (k = 16*hi + s)
    v16h a[16];
    {
        const _Float16* arow = xh + (size_t)(r0 + l) * DIM;
        #pragma unroll
        for (int kc = 0; kc < 16; ++kc) {
            FragU f;
            f.q[0] = *(const uint4*)(arow + kc * 32 + hi * 16);
            f.q[1] = *(const uint4*)(arow + kc * 32 + hi * 16 + 8);
            a[kc] = f.v;
        }
    }

    float acc[8];
    #pragma unroll
    for (int v = 0; v < 8; ++v) acc[v] = 0.f;

    // ---- prime the B pipeline: pair p = wave, chunks 0 and 1 of both tiles
    FragU b0[4], b1[4];
    {
        const _Float16* p0 = xh + ((size_t)(32 * wave) + l) * DIM;
        #pragma unroll
        for (int kc = 0; kc < 2; ++kc) {
            const int ko = kc * 32 + hi * 16;
            b0[kc].q[0] = *(const uint4*)(p0 + ko);
            b0[kc].q[1] = *(const uint4*)(p0 + ko + 8);
            b1[kc].q[0] = *(const uint4*)(p0 + PANEL + ko);
            b1[kc].q[1] = *(const uint4*)(p0 + PANEL + ko + 8);
        }
    }

    for (int t = 0; t < 16; ++t) {
        const int pidx = wave + 8 * t;
        const int n0 = 2 * pidx, n1 = n0 + 1;
        const _Float16* p0  = xh + ((size_t)(32 * pidx) + l) * DIM;
        const int tn = (t + 1) & 15;                       // wrap: t=15 reloads t=0
        const _Float16* np0 = xh + ((size_t)(32 * (wave + 8 * tn)) + l) * DIM;

        v8f c0 = {}, c1 = {};
        #pragma unroll
        for (int kc = 0; kc < 16; ++kc) {
            // distance-2 prefetch; rolls into next pair's chunks 0,1 at kc=14,15
            const int nb = (kc + 2) & 3;
            const _Float16* src = (kc < 14) ? p0 : np0;
            const int ko = ((kc < 14) ? (kc + 2) * 32 : (kc - 14) * 32) + hi * 16;
            b0[nb].q[0] = *(const uint4*)(src + ko);
            b0[nb].q[1] = *(const uint4*)(src + ko + 8);
            b1[nb].q[0] = *(const uint4*)(src + PANEL + ko);
            b1[nb].q[1] = *(const uint4*)(src + PANEL + ko + 8);

            c0 = wmma_f16(a[kc], b0[kc & 3].v, c0);
            c1 = wmma_f16(a[kc], b1[kc & 3].v, c1);
        }

        // warm next pair's deeper lines while we do the (tiny) tail
        __builtin_prefetch(np0 + 96, 0, 0);
        __builtin_prefetch(np0 + PANEL + 96, 0, 0);

        // ---- fixed-shift softmax accumulation (max := 2.0, provably safe)
        const bool d0 = (n0 == m_tile), d1 = (n1 == m_tile);
        #pragma unroll
        for (int v = 0; v < 8; ++v) {
            const int Mv = v + 8 * hi;                      // row within tile
            float s0 = INV_TEMP * c0[v];
            float s1 = INV_TEMP * c1[v];
            if (d0) {
                if (l == (Mv ^ 1)) t_sh[Mv] = s0;           // target logit
                if (l == Mv)       s0 = -20.f;              // diag -> exp ~ 0
            }
            if (d1) {
                if (l == (Mv ^ 1)) t_sh[Mv] = s1;
                if (l == Mv)       s1 = -20.f;
            }
            acc[v] += __expf(s0 - 2.0f) + __expf(s1 - 2.0f);
        }
    }

    // ---- one cross-lane reduction at the end (16-lane butterflies per row)
    #pragma unroll
    for (int v = 0; v < 8; ++v) {
        float s = acc[v];
        #pragma unroll
        for (int off = 1; off < 16; off <<= 1) s += __shfl_xor(s, off);
        if (l == 0) red_s[wave][v + 8 * hi] = s;
    }
    __syncthreads();

    // ---- cross-wave merge + per-strip loss (lanes 0..15 = rows)
    if (tid < 16) {
        float s = 0.f;
        #pragma unroll
        for (int w = 0; w < 8; ++w) s += red_s[w][tid];
        float li = (2.0f + __logf(s)) - t_sh[tid];          // -logp at target
        #pragma unroll
        for (int off = 1; off < 16; off <<= 1) li += __shfl_xor(li, off);
        if (tid == 0) partial[m_tile] = li;                 // one deterministic writer
    }
}

// ---------------------------------------------------------------------------
// Kernel 3: deterministic final reduction of 256 strip partials -> mean loss
// ---------------------------------------------------------------------------
__global__ void __launch_bounds__(256)
finalize_kernel(const float* __restrict__ partial, float* __restrict__ out) {
    const int tid = threadIdx.x;
    float v = partial[tid];
    #pragma unroll
    for (int off = 16; off > 0; off >>= 1) v += __shfl_xor(v, off);

    __shared__ float wsum[8];
    if ((tid & 31) == 0) wsum[tid >> 5] = v;
    __syncthreads();
    if (tid == 0) {
        float s = 0.f;
        #pragma unroll
        for (int w = 0; w < 8; ++w) s += wsum[w];
        out[0] = s * (1.0f / (float)N_ROWS);
    }
}

// ---------------------------------------------------------------------------
extern "C" void kernel_launch(void* const* d_in, const int* in_sizes, int n_in,
                              void* d_out, int out_size, void* d_ws, size_t ws_size,
                              hipStream_t stream) {
    const float* x   = (const float*)d_in[0];
    float*       out = (float*)d_out;

    // workspace layout: [0,1KB): 256 strip partials | [1KB, 1KB+4MB): xh f16
    float*    partial = (float*)d_ws;
    _Float16* xh      = (_Float16*)((char*)d_ws + 1024);

    normalize_kernel   <<<N_ROWS,  256, 0, stream>>>(x, xh);
    ntxent_strip_kernel<<<N_TILES, 256, 0, stream>>>(xh, partial);
    finalize_kernel    <<<1,       256, 0, stream>>>(partial, out);
}
